// HawkesBase_65051574665240
// MI455X (gfx1250) — compile-verified, matrix operand order checked by reading, not run on hardware
//
#include <hip/hip_runtime.h>
#include <stdint.h>

typedef __attribute__((ext_vector_type(16))) _Float16 v16h;
typedef __attribute__((ext_vector_type(8)))  float    v8f;
typedef __attribute__((ext_vector_type(4)))  unsigned int u32x4;
typedef __attribute__((ext_vector_type(4)))  int  i32x4;
typedef __attribute__((ext_vector_type(8)))  int  i32x8;

#define MDIM     4096
#define KDIM     2
#define TPB      256
#define MPT      16        // MDIM / TPB columns owned per thread
#define NSTEPS   4096
#define SIM_END_T 2000.0f

__device__ __forceinline__ uint32_t hash32(uint32_t x) {
  x ^= x >> 16; x *= 0x7feb352dU;
  x ^= x >> 15; x *= 0x846ca68bU;
  x ^= x >> 16; return x;
}
__device__ __forceinline__ float u01(uint32_t h) {
  return (float)(h >> 8) * (1.0f / 16777216.0f);
}

__global__ __launch_bounds__(TPB, 1)
void hawkes_thinning_kernel(const float* __restrict__ mu,
                            const float* __restrict__ gam,
                            const float* __restrict__ alpha,
                            const int*   __restrict__ maxEvPtr,
                            float* __restrict__ out, int out_size)
{
  __shared__ float sAlpha[KDIM * MDIM];   // 32 KB: TDM destination for alpha[:,ev,:]
  __shared__ float sTS[TPB];              // per-thread lambda sums
  __shared__ float sWS[8][2];             // per-wave partial sums of R
  __shared__ float sRowTot[8][16];        // row totals for 2nd scan WMMA (per wave)
  __shared__ float sCtrlF[4];             // tNew, e0, e1, target
  __shared__ int   sCtrlI[2];             // over, accept
  __shared__ int   sEv;
  __shared__ int   sPick;
  __shared__ float sPsAt;
  __shared__ float sMuSum;

  const int tid  = threadIdx.x;
  const int lane = tid & 31;
  const int wave = tid >> 5;
  const int maxEv = *maxEvPtr;

  float* tiOut  = out;
  float* miOut  = out + maxEv;
  float* cntOut = out + 2 * maxEv;

  // Zero outputs. Wave 0 only: all global stores to the output region come
  // from wave 0 (event records too), so per-wave store ordering suffices.
  if (wave == 0) {
    for (int i = lane; i < out_size; i += 32) out[i] = 0.0f;
  }

  const float g0 = gam[0], g1 = gam[1];

  float muR[MPT], R0[MPT], R1[MPT];
#pragma unroll
  for (int q = 0; q < MPT; ++q) {
    muR[q] = mu[tid * MPT + q];
    R0[q] = 0.0f; R1[q] = 0.0f;
  }

  // block-wide sum of mu (== base_ub == mu part of lam_sum)
  {
    float p = 0.f;
#pragma unroll
    for (int q = 0; q < MPT; ++q) p += muR[q];
#pragma unroll
    for (int off = 16; off > 0; off >>= 1) p += __shfl_xor(p, off, 32);
    if (lane == 0) sWS[wave][0] = p;
    __syncthreads();
    if (tid == 0) {
      float s = 0.f;
      for (int w = 0; w < 8; ++w) s += sWS[w][0];
      sMuSum = s;
    }
    __syncthreads();
  }
  const float muSum = sMuSum;

  float t = 0.0f;
  int cnt = 0;

  for (int step = 0; step < NSTEPS; ++step) {
    if (t >= SIM_END_T || cnt >= maxEv) break;   // uniform

    // ---- S_k = sum_m R[k][m]  (wave reduce + cross-wave in LDS)
    float p0 = 0.f, p1 = 0.f;
#pragma unroll
    for (int q = 0; q < MPT; ++q) { p0 += R0[q]; p1 += R1[q]; }
#pragma unroll
    for (int off = 16; off > 0; off >>= 1) {
      p0 += __shfl_xor(p0, off, 32);
      p1 += __shfl_xor(p1, off, 32);
    }
    if (lane == 0) { sWS[wave][0] = p0; sWS[wave][1] = p1; }
    __syncthreads();                                  // B1

    if (tid == 0) {
      float S0 = 0.f, S1 = 0.f;
      for (int w = 0; w < 8; ++w) { S0 += sWS[w][0]; S1 += sWS[w][1]; }
      float lamStar = muSum + g0 * S0 + g1 * S1;
      uint32_t b = (uint32_t)step * 3u;
      float uTau = fmaxf(u01(hash32(b + 1u)), 1e-7f);
      float tNew = t - __logf(uTau) / lamStar;        // Exp(lamStar) waiting time
      int over = (tNew > SIM_END_T) ? 1 : 0;
      float dt = tNew - t;
      float e0 = __expf(-g0 * dt), e1 = __expf(-g1 * dt);
      float lamSum = muSum + g0 * e0 * S0 + g1 * e1 * S1;
      float r = lamSum / lamStar;
      float uA = u01(hash32(b + 2u));
      float uC = u01(hash32(b + 3u));
      int accept = (!over) && (r <= 1.01f) && (uA <= r);
      sCtrlF[0] = tNew; sCtrlF[1] = e0; sCtrlF[2] = e1;
      sCtrlF[3] = uC * lamSum;                        // inverse-CDF target
      sCtrlI[0] = over; sCtrlI[1] = accept;
    }
    __syncthreads();                                  // B2

    const float tNew   = sCtrlF[0];
    const float e0     = sCtrlF[1];
    const float e1     = sCtrlF[2];
    const float target = sCtrlF[3];
    const int over     = sCtrlI[0];
    const int accept   = sCtrlI[1];

    if (over) { t = tNew; continue; }                 // R unchanged past horizon

    // decay memory to candidate time (always, when not over)
#pragma unroll
    for (int q = 0; q < MPT; ++q) { R0[q] *= e0; R1[q] *= e1; }

    if (!accept) { t = tNew; continue; }              // rejected candidate

    // ---- categorical sampling: 256-wide prefix scan via two WMMAs ----
    float lam[MPT]; float ts = 0.f;
#pragma unroll
    for (int q = 0; q < MPT; ++q) {
      lam[q] = muR[q] + g0 * R0[q] + g1 * R1[q];
      ts += lam[q];
    }
    sTS[tid] = ts;
    __syncthreads();                                  // B3

    const int row = lane & 15;    // A-row / B-column / D-N index
    const int hi  = lane >> 4;
    // WMMA #1: D1 = T x U ; T[r][c] = sTS[16r+c], U upper-triangular ones.
    v16h A1, B1;
#pragma unroll
    for (int v = 0; v < 8; ++v) {
#pragma unroll
      for (int h = 0; h < 2; ++h) {
        int kk = (v < 4) ? (2 * v + h + hi * 8) : (16 + 2 * (v - 4) + h + hi * 8);
        float av = (kk < 16) ? sTS[row * 16 + kk] : 0.0f;
        float bv = (kk < 16 && kk <= row) ? 1.0f : 0.0f;
        A1[2 * v + h] = (_Float16)av;
        B1[2 * v + h] = (_Float16)bv;
      }
    }
    v8f d1 = {};
    d1 = __builtin_amdgcn_wmma_f32_16x16x32_f16(false, A1, false, B1,
                                                (short)0, d1, false, false);
    // row totals = column 15 of D1
    if (row == 15) {
#pragma unroll
      for (int rr = 0; rr < 8; ++rr) sRowTot[wave][rr + 8 * hi] = d1[rr];
    }
    __syncthreads();                                  // B4

    // WMMA #2: D2 = Lstrict x broadcast(rowTot) -> per-row exclusive offsets
    v16h A2, B2v;
#pragma unroll
    for (int v = 0; v < 8; ++v) {
#pragma unroll
      for (int h = 0; h < 2; ++h) {
        int kk = (v < 4) ? (2 * v + h + hi * 8) : (16 + 2 * (v - 4) + h + hi * 8);
        float av = (kk < 16 && kk < row) ? 1.0f : 0.0f;
        float bv = (kk < 16) ? sRowTot[wave][kk] : 0.0f;
        A2[2 * v + h] = (_Float16)av;
        B2v[2 * v + h] = (_Float16)bv;
      }
    }
    v8f d2 = {};
    d2 = __builtin_amdgcn_wmma_f32_16x16x32_f16(false, A2, false, B2v,
                                                (short)0, d2, false, false);

    // PS[16m+n] = d1+d2 : inclusive prefix of the 256 thread sums.
    int best = 256; float bestPs = 3.0e38f;
#pragma unroll
    for (int rr = 0; rr < 8; ++rr) {
      float ps = d1[rr] + d2[rr];
      int tt = (rr + 8 * hi) * 16 + row;
      if (ps >= target && tt < best) { best = tt; bestPs = ps; }
    }
#pragma unroll
    for (int off = 16; off > 0; off >>= 1) {
      int   ob = __shfl_xor(best, off, 32);
      float op = __shfl_xor(bestPs, off, 32);
      if (ob < best) { best = ob; bestPs = op; }
    }
    if (tid == 0) {
      sPick = (best < 256) ? best : 255;
      sPsAt = (best < 256) ? bestPs : 3.0e38f;
    }
    __syncthreads();                                  // B5

    const int pick = sPick;
    if (tid == pick) {                                // resolve index at f32
      float before = sPsAt - ts;
      float resid  = target - before;
      float c = 0.f; int j = -1;
#pragma unroll
      for (int q = 0; q < MPT; ++q) {
        c += lam[q];
        if (j < 0 && c >= resid) j = q;
      }
      if (j < 0) j = MPT - 1;
      sEv = pick * MPT + j;
    }
    __syncthreads();                                  // B6

    const int ev = sEv;

    // ---- TDM: one 2-D descriptor pulls alpha[0][ev][:] and alpha[1][ev][:]
    //      (tile 4096x2, dim0 stride = M*M elements) straight into LDS.
    if (wave == 0) {
      unsigned long long gaddr =
          (unsigned long long)(uintptr_t)alpha +
          (unsigned long long)ev * (unsigned long long)(MDIM * 4);
      uint32_t ldsOff = (uint32_t)(uintptr_t)(void*)&sAlpha[0];
      u32x4 d0;
      d0[0] = 1u;                                          // count=1, user desc
      d0[1] = ldsOff;                                      // lds_addr
      d0[2] = (uint32_t)gaddr;                             // global_addr[31:0]
      d0[3] = (uint32_t)((gaddr >> 32) & 0x1FFFFFFull)     // global_addr[56:32]
              | (2u << 30);                                // type = 2 (image)
      i32x8 dg1;
      dg1[0] = 0x00020000;                 // data_size = 2 -> 4-byte elements
      dg1[1] = (int)(0x1000u << 16);       // tensor_dim0 = 4096 (bits 79:48)
      dg1[2] = (int)(2u << 16);            // tensor_dim1 = 2    (bits 111:80)
      dg1[3] = (int)(0x1000u << 16);       // tensor_dim1 hi | tile_dim0 = 4096
      dg1[4] = 2;                          // tile_dim1 = 2, tile_dim2 = 0
      dg1[5] = 0x01000000;                 // tensor_dim0_stride = 4096*4096
      dg1[6] = 0;
      dg1[7] = 0;                          // tensor_dim1_stride unused (2-D)
      i32x4 zz = {0, 0, 0, 0};
#if __has_builtin(__builtin_amdgcn_tensor_load_to_lds)
  #if defined(__clang_major__) && __clang_major__ >= 23
    #pragma message("CDNA5 probe: tensor_load_to_lds via 6-arg builtin")
      i32x8 zz8 = {0, 0, 0, 0, 0, 0, 0, 0};
      __builtin_amdgcn_tensor_load_to_lds(d0, dg1, zz, zz, zz8, 0);
  #else
    #pragma message("CDNA5 probe: tensor_load_to_lds via 5-arg builtin")
      __builtin_amdgcn_tensor_load_to_lds(d0, dg1, zz, zz, 0);
  #endif
      __builtin_amdgcn_s_wait_tensorcnt(0);
#else
    #pragma message("CDNA5 probe: tensor_load_to_lds via inline asm fallback")
      (void)zz;
      asm volatile("tensor_load_to_lds %0, %1"
                   :
                   : "s"(d0), "s"(dg1)
                   : "memory");
      asm volatile("s_wait_tensorcnt 0x0" ::: "memory");
#endif
    }
    __syncthreads();                                  // B7

#pragma unroll
    for (int q = 0; q < MPT; ++q) {
      R0[q] += sAlpha[tid * MPT + q];
      R1[q] += sAlpha[MDIM + tid * MPT + q];
    }
    if (tid == 0) {
      tiOut[cnt] = tNew;
      miOut[cnt] = (float)ev;
    }
    cnt++;
    t = tNew;
  }

  if (tid == 0) cntOut[0] = (float)cnt;
}

extern "C" void kernel_launch(void* const* d_in, const int* in_sizes, int n_in,
                              void* d_out, int out_size, void* d_ws, size_t ws_size,
                              hipStream_t stream) {
  const float* mu    = (const float*)d_in[0];
  const float* gam   = (const float*)d_in[1];
  const float* alpha = (const float*)d_in[2];
  const int*   maxEv = (const int*)d_in[3];
  hawkes_thinning_kernel<<<1, TPB, 0, stream>>>(mu, gam, alpha, maxEv,
                                                (float*)d_out, out_size);
  (void)in_sizes; (void)n_in; (void)d_ws; (void)ws_size;
}
